// Mamba2_LM_35321811042485
// MI455X (gfx1250) — compile-verified, matrix operand order checked by reading, not run on hardware
//
#include <hip/hip_runtime.h>

// ---------------- problem constants ----------------
constexpr int Hh    = 32;                       // heads
constexpr int KVHn  = 8;                        // kv heads
constexpr int Pp    = 64;                       // head dim / state dim
constexpr int Dd    = Hh * Pp;                  // 2048
constexpr int CONVD = (Hh + 2 * KVHn) * Pp;     // 3072
constexpr int PROJn = (Hh + 2 * KVHn + Hh) * Pp + Hh; // 5152
constexpr int Bb    = 2;
constexpr int Ll    = 2048;
constexpr int BL    = Bb * Ll;                  // 4096
constexpr int CHK   = 128;
constexpr int NC    = Ll / CHK;                 // 16 chunks

typedef __attribute__((ext_vector_type(16))) __bf16 v16bf;
typedef __attribute__((ext_vector_type(8)))  float  v8f;
typedef __attribute__((ext_vector_type(8)))  __bf16 bf16x8;
typedef __attribute__((ext_vector_type(4)))  unsigned int u32x4;
typedef __attribute__((ext_vector_type(8)))  int i32x8;
typedef __attribute__((ext_vector_type(4)))  int i32x4;

// ---------------- helpers ----------------
__device__ __forceinline__ v8f wmma_bf16(v16bf a, v16bf b, v8f c) {
  return __builtin_amdgcn_wmma_f32_16x16x32_bf16(false, a, false, b, (short)0, c,
                                                 false, false);
}

// Build a 16x16x32 A/B fragment from a row-major bf16 row (LDS or global).
// frag[0..7]  = row[base+0..7], frag[8..15] = row[base+16..23], base = 8*hi.
__device__ __forceinline__ v16bf make_frag(const __bf16* lo, const __bf16* hi) {
  bf16x8 a = *(const bf16x8*)lo;
  bf16x8 b = *(const bf16x8*)hi;
  v16bf r;
#pragma unroll
  for (int e = 0; e < 8; ++e) { r[e] = a[e]; r[8 + e] = b[e]; }
  return r;
}

// Same fragment, sourced from an f32 row (LDS or global), converted to bf16.
__device__ __forceinline__ v16bf frag_from_f32(const float* p, int off) {
  float4 a = *(const float4*)(p + off);
  float4 b = *(const float4*)(p + off + 4);
  float4 c = *(const float4*)(p + off + 16);
  float4 d = *(const float4*)(p + off + 20);
  v16bf r;
  r[0]  = (__bf16)a.x; r[1]  = (__bf16)a.y; r[2]  = (__bf16)a.z; r[3]  = (__bf16)a.w;
  r[4]  = (__bf16)b.x; r[5]  = (__bf16)b.y; r[6]  = (__bf16)b.z; r[7]  = (__bf16)b.w;
  r[8]  = (__bf16)c.x; r[9]  = (__bf16)c.y; r[10] = (__bf16)c.z; r[11] = (__bf16)c.w;
  r[12] = (__bf16)d.x; r[13] = (__bf16)d.y; r[14] = (__bf16)d.z; r[15] = (__bf16)d.w;
  return r;
}

__device__ __forceinline__ unsigned lds_offset(const void* p) {
  // flat LDS address: low 32 bits are the LDS byte offset
  return (unsigned)(uintptr_t)p;
}

// Issue a TDM 2D tile load: 32 f32 wide x `rows`, source row stride
// `stride_elems` f32, into LDS at byte offset `lds_off`, with 4-DWORD LDS
// padding after every 32 DWORDs (dest row stride = 36 f32 = 144B).
__device__ __forceinline__ void tdm_load_2d(const float* gsrc, unsigned lds_off,
                                            int rows, int stride_elems) {
  u32x4 g0;
  unsigned long long ga = (unsigned long long)(uintptr_t)gsrc;
  g0[0] = 1u;                                            // count=1 (user D#)
  g0[1] = lds_off;                                       // lds_addr (bytes)
  g0[2] = (unsigned)(ga & 0xffffffffu);                  // global_addr[31:0]
  g0[3] = (unsigned)((ga >> 32) & 0x01ffffffu) | 0x80000000u; // addr[56:32]|type=2
  i32x8 g1;
  g1[0] = (int)((2u << 16)       // data_size = 4 bytes
              | (1u << 20)       // pad_enable
              | (4u << 22)       // pad_interval = 32 DWORDs
              | (3u << 25));     // pad_amount  = 4 DWORDs
  g1[1] = (int)((32u & 0xffffu) << 16);                  // tensor_dim0 = 32 (lo16)
  g1[2] = (int)(((unsigned)rows & 0xffffu) << 16);       // tdim0 hi | tensor_dim1 lo
  g1[3] = (int)(32u << 16);                              // tdim1 hi | tile_dim0 = 32
  g1[4] = (int)((unsigned)rows & 0xffffu);               // tile_dim1 = rows
  g1[5] = (int)(unsigned)stride_elems;                   // tensor_dim0_stride lo32
  g1[6] = 0;                                             // stride hi | dim1_stride lo
  g1[7] = 0;
  i32x4 z4 = {0, 0, 0, 0};
#if defined(__clang_major__) && __clang_major__ >= 23
  i32x8 z8 = {0, 0, 0, 0, 0, 0, 0, 0};
  __builtin_amdgcn_tensor_load_to_lds(g0, g1, z4, z4, z8, 0);
#else
  __builtin_amdgcn_tensor_load_to_lds(g0, g1, z4, z4, 0);
#endif
}

// ---- kernel 1/8: C[M,N] = A[M,K] * Bw[N,K]^T  (TDM double-buffer, bf16 WMMA)
__global__ __launch_bounds__(256)
void gemm_nt_bf16(const float* __restrict__ A, const float* __restrict__ Bw,
                  float* __restrict__ C, int M, int N, int K) {
  constexpr int KF = 36;                         // padded f32 row stride (144B)
  __shared__ __align__(16) float As[2][64 * KF];
  __shared__ __align__(16) float Bs[2][128 * KF];
  const int t    = threadIdx.x;
  const int lane = t & 31, wvv = t >> 5;
  const int wm   = wvv >> 2, wn = wvv & 3;
  const int m0   = blockIdx.x * 64;
  const int n0   = blockIdx.y * 128;
  const int mr   = lane & 15, hi = lane >> 4;
  const int nrows = (N - n0 < 128) ? (N - n0) : 128;
  v8f acc[2][2] = {};
  if (t == 0) {                                  // prologue: tile k0=0
    tdm_load_2d(A  + (size_t)m0 * K, lds_offset(&As[0][0]), 64, K);
    tdm_load_2d(Bw + (size_t)n0 * K, lds_offset(&Bs[0][0]), nrows, K);
  }
  if (wvv == 0) __builtin_amdgcn_s_wait_tensorcnt(0);
  __syncthreads();
  for (int k0 = 0; k0 < K; k0 += 32) {
    const int cur = (k0 >> 5) & 1;
    if (t == 0 && k0 + 32 < K) {                 // prefetch next tile via TDM
      tdm_load_2d(A  + (size_t)m0 * K + k0 + 32, lds_offset(&As[cur ^ 1][0]), 64, K);
      tdm_load_2d(Bw + (size_t)n0 * K + k0 + 32, lds_offset(&Bs[cur ^ 1][0]), nrows, K);
    }
    v16bf bfr[2];
#pragma unroll
    for (int j = 0; j < 2; ++j)
      bfr[j] = frag_from_f32(&Bs[cur][(32 * wn + 16 * j + mr) * KF], 8 * hi);
#pragma unroll
    for (int i = 0; i < 2; ++i) {
      v16bf af = frag_from_f32(&As[cur][(32 * wm + 16 * i + mr) * KF], 8 * hi);
#pragma unroll
      for (int j = 0; j < 2; ++j)
        acc[i][j] = wmma_bf16(af, bfr[j], acc[i][j]);
    }
    if (wvv == 0 && k0 + 32 < K) __builtin_amdgcn_s_wait_tensorcnt(0);
    __syncthreads();
  }
#pragma unroll
  for (int i = 0; i < 2; ++i)
#pragma unroll
    for (int j = 0; j < 2; ++j)
#pragma unroll
      for (int v = 0; v < 8; ++v) {
        int row = m0 + 32 * wm + 16 * i + v + 8 * hi;
        int col = n0 + 32 * wn + 16 * j + mr;
        if (col < N) C[(size_t)row * N + col] = acc[i][j][v];
      }
}

// ---------------- kernel 2/8: depthwise conv(k=2)+SiLU, softplus dt, A*dt ----
__global__ __launch_bounds__(256)
void conv_silu_dt(const float* __restrict__ proj, const float* __restrict__ conv_w,
                  const float* __restrict__ conv_b, const float* __restrict__ dt_bias,
                  const float* __restrict__ A_log_bias,
                  float* __restrict__ vkq, float* __restrict__ dt_out,
                  float* __restrict__ adt_out) {
  const int bl = blockIdx.x;
  const int l  = bl & (Ll - 1);
  const float* cur = proj + (size_t)bl * PROJn;
  const float* prv = cur - PROJn;
  float* out = vkq + (size_t)bl * CONVD;
  for (int c = threadIdx.x; c < CONVD; c += 256) {
    float xp = (l > 0) ? prv[c] : 0.f;
    float s  = xp * conv_w[2 * c] + cur[c] * conv_w[2 * c + 1] + conv_b[c];
    out[c]   = s / (1.f + expf(-s));
  }
  if (threadIdx.x < Hh) {
    int h = threadIdx.x;
    float raw = cur[CONVD + Dd + h] + dt_bias[h];
    float dtv = (raw > 20.f) ? raw : log1pf(expf(raw));
    dt_out[(size_t)bl * Hh + h] = dtv;
    float Ah = -expf(A_log_bias[h]);
    int b = bl >> 11;                            // / Ll
    adt_out[((size_t)(b * Hh + h)) * Ll + l] = dtv * Ah;
  }
}

// ---------------- kernel 3/8: per-chunk inclusive cumsum of A*dt ------------
__global__ __launch_bounds__(128)
void chunk_scan(const float* __restrict__ adt, float* __restrict__ Acs,
                float* __restrict__ alast) {
  __shared__ float s[CHK];
  const int bh = blockIdx.x;
  const int t  = threadIdx.x;
  for (int c = 0; c < NC; ++c) {
    const size_t base = (size_t)bh * Ll + c * CHK;
    s[t] = adt[base + t];
    __syncthreads();
    for (int off = 1; off < CHK; off <<= 1) {
      float v = (t >= off) ? s[t - off] : 0.f;
      __syncthreads();
      s[t] += v;
      __syncthreads();
    }
    Acs[base + t] = s[t];
    if (t == CHK - 1) alast[bh * NC + c] = s[CHK - 1];
    __syncthreads();
  }
}

// ---------------- kernel 4/8: intra-chunk SSD + chunk-end states -----------
__global__ __launch_bounds__(256)
void ssd_intra(const float* __restrict__ vkq, const float* __restrict__ dt,
               const float* __restrict__ Acs, const float* __restrict__ alast,
               float* __restrict__ y, float* __restrict__ st) {
  constexpr int MP = 136;                        // 128 + 8 pad (bf16)
  __shared__ __align__(16) __bf16 Ms[128 * MP];  // masked+decayed q.k^T
  __shared__ __align__(16) __bf16 xcT[64 * 128]; // x^T: [p][l], x = v*dt
  __shared__ float Acs_s[CHK];
  __shared__ float wl[CHK];                      // exp(a_last - Acs[l])
  const int blk = blockIdx.x;
  const int c = blk & 15, bh = blk >> 4;
  const int b = bh >> 5, h = bh & 31;
  const int t = threadIdx.x, lane = t & 31, wv = t >> 5;
  const int mr = lane & 15, hi = lane >> 4;
  const size_t bl0 = (size_t)b * Ll + c * CHK;   // base row in [B*L]
  const int kv = h >> 2;                         // kv head
  const float al = alast[bh * NC + c];
  if (t < CHK) {
    float a = Acs[(size_t)bh * Ll + c * CHK + t];
    Acs_s[t] = a;
    wl[t]    = expf(al - a);
  }
  for (int i = t; i < 128 * MP; i += 256) Ms[i] = (__bf16)0.f;
  for (int i = t; i < 64 * 128; i += 256) {      // coalesced over p
    int p = i & 63, l = i >> 6;
    float vv = vkq[(bl0 + l) * CONVD + kv * Pp + p];
    float dd = dt[(bl0 + l) * Hh + h];
    xcT[p * 128 + l] = (__bf16)(vv * dd);
  }
  __syncthreads();
  { // S = q.k^T, lower-triangular tiles only; scale by exp(Acs[i]-Acs[j])
    const int i0 = 16 * wv;
    for (int jt = 0; jt <= wv; ++jt) {
      v8f acc = {};
#pragma unroll
      for (int ks = 0; ks < 2; ++ks) {
        const int n0 = ks * 32;
        const float* qrow =
            vkq + (bl0 + i0 + mr) * CONVD + 2 * KVHn * Pp + h * Pp + n0;
        v16bf af = frag_from_f32(qrow, 8 * hi);
        const float* krow =
            vkq + (bl0 + 16 * jt + mr) * CONVD + KVHn * Pp + kv * Pp + n0;
        v16bf bf = frag_from_f32(krow, 8 * hi);
        acc = wmma_bf16(af, bf, acc);
      }
#pragma unroll
      for (int v = 0; v < 8; ++v) {
        int irow = i0 + v + 8 * hi;
        int jcol = 16 * jt + mr;
        float sc = (jcol <= irow) ? expf(Acs_s[irow] - Acs_s[jcol]) : 0.f;
        Ms[irow * MP + jcol] = (__bf16)(acc[v] * sc);
      }
    }
  }
  __syncthreads();
  { // Y_diag = Ms @ x   (wave -> one 16-row strip, all 64 cols)
    const int i0 = 16 * wv;
    v8f accp[4] = {};
#pragma unroll
    for (int ls = 0; ls < 4; ++ls) {
      const int l0 = ls * 32;
      const __bf16* arow = Ms + (i0 + mr) * MP + l0;
      v16bf af = make_frag(arow + 8 * hi, arow + 16 + 8 * hi);
#pragma unroll
      for (int pt = 0; pt < 4; ++pt) {
        const __bf16* brow = xcT + (16 * pt + mr) * 128 + l0;
        v16bf bf = make_frag(brow + 8 * hi, brow + 16 + 8 * hi);
        accp[pt] = wmma_bf16(af, bf, accp[pt]);
      }
    }
#pragma unroll
    for (int pt = 0; pt < 4; ++pt)
#pragma unroll
      for (int v = 0; v < 8; ++v) {
        int irow = i0 + v + 8 * hi;
        y[((bl0 + irow) * Hh + h) * Pp + 16 * pt + mr] = accp[pt][v];
      }
  }
  { // chunk-end state st[p][n] = sum_l k[l,n]*wl[l]*x[l,p]
#pragma unroll
    for (int u = 0; u < 2; ++u) {
      int tile = 2 * wv + u;
      int nt = tile >> 2, pt = tile & 3;
      v8f acc = {};
      for (int ls = 0; ls < 4; ++ls) {
        int l0 = ls * 32;
        v16bf af;
        int n = 16 * nt + mr;
        const float* kcol = vkq + KVHn * Pp + kv * Pp + n;
#pragma unroll
        for (int e = 0; e < 8; ++e) {
          int l1 = l0 + 8 * hi + e;
          int l2 = l0 + 16 + 8 * hi + e;
          af[e]     = (__bf16)(kcol[(bl0 + l1) * CONVD] * wl[l1]);
          af[8 + e] = (__bf16)(kcol[(bl0 + l2) * CONVD] * wl[l2]);
        }
        const __bf16* brow = xcT + (16 * pt + mr) * 128 + l0;
        v16bf bf = make_frag(brow + 8 * hi, brow + 16 + 8 * hi);
        acc = wmma_bf16(af, bf, acc);
      }
#pragma unroll
      for (int v = 0; v < 8; ++v) {
        int n = 16 * nt + v + 8 * hi;
        int p = 16 * pt + mr;
        st[(((size_t)blk) * 64 + p) * 64 + n] = acc[v];
      }
    }
  }
}

// ---------------- kernel 5/8: sequential chunk-state recurrence ------------
__global__ __launch_bounds__(256)
void state_scan(const float* __restrict__ st, const float* __restrict__ alast,
                float* __restrict__ Rg) {
  const int bh = blockIdx.x;
  const int t  = threadIdx.x;
  float r[16];
#pragma unroll
  for (int e = 0; e < 16; ++e) r[e] = 0.f;
  const size_t base = (size_t)bh * NC * 4096;
  for (int z = 0; z < NC; ++z) {
    float* out = Rg + base + (size_t)z * 4096 + t * 16;
#pragma unroll
    for (int e = 0; e < 16; ++e) out[e] = r[e];
    const float ea = expf(alast[bh * NC + z]);
    const float* s = st + base + (size_t)z * 4096 + t * 16;
#pragma unroll
    for (int e = 0; e < 16; ++e) r[e] = r[e] * ea + s[e];
  }
}

// ---------------- kernel 6/8: inter-chunk contribution Y += exp(Acs)*q.R ---
__global__ __launch_bounds__(256)
void ssd_inter(const float* __restrict__ vkq, const float* __restrict__ Acs,
               const float* __restrict__ Rg, float* __restrict__ y) {
  const int blk = blockIdx.x;
  const int c = blk & 15, bh = blk >> 4;
  const int b = bh >> 5, h = bh & 31;
  const int t = threadIdx.x, lane = t & 31, wv = t >> 5;
  const int mr = lane & 15, hi = lane >> 4;
  const size_t bl0 = (size_t)b * Ll + c * CHK;
  const int i0 = 16 * wv;
  const float* Rc = Rg + ((size_t)blk) * 4096;   // [p][n] row-major
  v8f accp[4] = {};
#pragma unroll
  for (int ks = 0; ks < 2; ++ks) {
    int n0 = ks * 32;
    const float* qrow =
        vkq + (bl0 + i0 + mr) * CONVD + 2 * KVHn * Pp + h * Pp + n0;
    v16bf af = frag_from_f32(qrow, 8 * hi);
#pragma unroll
    for (int pt = 0; pt < 4; ++pt) {
      const float* rrow = Rc + (16 * pt + mr) * 64 + n0;
      v16bf bf = frag_from_f32(rrow, 8 * hi);
      accp[pt] = wmma_bf16(af, bf, accp[pt]);
    }
  }
#pragma unroll
  for (int v = 0; v < 8; ++v) {
    int irow = i0 + v + 8 * hi;
    float ea = expf(Acs[(size_t)bh * Ll + c * CHK + irow]);
#pragma unroll
    for (int pt = 0; pt < 4; ++pt) {
      size_t idx = ((bl0 + irow) * Hh + h) * Pp + 16 * pt + mr;
      y[idx] += ea * accp[pt][v];
    }
  }
}

// ---------------- kernel 7/8: RMSNorm(P) * gnorm_w * SiLU(gate) -------------
__global__ __launch_bounds__(256)
void gate_norm(const float* __restrict__ y, const float* __restrict__ proj,
               const float* __restrict__ gnorm_w, float* __restrict__ act) {
  __shared__ float red[256];
  const int bl = blockIdx.x;
  const int t  = threadIdx.x;
  const int hh = t >> 3, sub = t & 7;            // 8 threads x 8 elems per head
  const float* yrow = y + (size_t)bl * Dd + hh * Pp + sub * 8;
  float v[8];
  float ss = 0.f;
#pragma unroll
  for (int e = 0; e < 8; ++e) { v[e] = yrow[e]; ss += v[e] * v[e]; }
  red[t] = ss;
  __syncthreads();
  float tot = 0.f;
#pragma unroll
  for (int e = 0; e < 8; ++e) tot += red[hh * 8 + e];
  float inv = rsqrtf(tot * (1.f / 64.f) + 1e-5f);
  const float* grow = proj + (size_t)bl * PROJn + CONVD + hh * Pp + sub * 8;
  float* orow = act + (size_t)bl * Dd + hh * Pp + sub * 8;
#pragma unroll
  for (int e = 0; e < 8; ++e) {
    float g  = grow[e];
    float sg = g / (1.f + expf(-g));
    orow[e]  = v[e] * inv * gnorm_w[sub * 8 + e] * sg;
  }
}

// ---------------- launcher -------------------------------------------------
extern "C" void kernel_launch(void* const* d_in, const int* in_sizes, int n_in,
                              void* d_out, int out_size, void* d_ws, size_t ws_size,
                              hipStream_t stream) {
  (void)in_sizes; (void)n_in; (void)out_size; (void)ws_size;
  const float* X        = (const float*)d_in[0];  // [B,L,D]
  const float* W1       = (const float*)d_in[1];  // [PROJ,D]
  const float* conv_w   = (const float*)d_in[2];  // [CONVD,2]
  const float* conv_b   = (const float*)d_in[3];  // [CONVD]
  const float* dt_bias  = (const float*)d_in[4];  // [H]
  const float* A_log    = (const float*)d_in[5];  // [H]
  const float* gw       = (const float*)d_in[6];  // [P]
  const float* Wo       = (const float*)d_in[7];  // [D,D]
  float* out = (float*)d_out;

  char* ws = (char*)d_ws;
  size_t off = 0;
  auto alloc = [&](size_t nfloats) {
    float* p = (float*)(ws + off);
    off += ((nfloats * sizeof(float) + 255) / 256) * 256;
    return p;
  };
  float* proj = alloc((size_t)BL * PROJn);        // 84.4 MB
  float* vkq  = alloc((size_t)BL * CONVD);        // 50.3 MB
  float* dtb  = alloc((size_t)BL * Hh);
  float* adt  = alloc((size_t)Bb * Hh * Ll);      // [bh][l]
  float* AcsB = alloc((size_t)Bb * Hh * Ll);      // [bh][l]
  float* alst = alloc((size_t)Bb * Hh * NC);
  float* yb   = alloc((size_t)BL * Dd);           // 33.5 MB
  float* stb  = alloc((size_t)Bb * Hh * NC * 64 * 64);  // 16.8 MB
  float* Rg   = alloc((size_t)Bb * Hh * NC * 64 * 64);  // 16.8 MB
  float* actb = alloc((size_t)BL * Dd);           // 33.5 MB

  // 1) in-projection: proj = X @ W1^T
  gemm_nt_bf16<<<dim3(BL / 64, (PROJn + 127) / 128), 256, 0, stream>>>(
      X, W1, proj, BL, PROJn, Dd);
  // 2) conv + SiLU + dt
  conv_silu_dt<<<BL, 256, 0, stream>>>(proj, conv_w, conv_b, dt_bias, A_log,
                                       vkq, dtb, adt);
  // 3) per-chunk cumsum of A*dt
  chunk_scan<<<Bb * Hh, 128, 0, stream>>>(adt, AcsB, alst);
  // 4) intra-chunk SSD + chunk-end states
  ssd_intra<<<Bb * Hh * NC, 256, 0, stream>>>(vkq, dtb, AcsB, alst, yb, stb);
  // 5) state recurrence across chunks
  state_scan<<<Bb * Hh, 256, 0, stream>>>(stb, alst, Rg);
  // 6) inter-chunk contribution
  ssd_inter<<<Bb * Hh * NC, 256, 0, stream>>>(vkq, AcsB, Rg, yb);
  // 7) RMSNorm + swish gate
  gate_norm<<<BL, 256, 0, stream>>>(yb, proj, gw, actb);
  // 8) out-projection: out = act @ Wo^T
  gemm_nt_bf16<<<dim3(BL / 64, Dd / 128), 256, 0, stream>>>(
      actb, Wo, out, BL, Dd, Dd);
}
